// HypConvLayer_83915071030023
// MI455X (gfx1250) — matrix operand order repaired and checked by reading, not compiled, not run on hardware
//
#include <hip/hip_runtime.h>
#include <hip/hip_bf16.h>
#include <cstdint>

// ---------------------------------------------------------------------------
// CDNA5 / gfx1250 fused Hyperbolic GCN layer (HGCN HypConv).
//  kernel 1: 850k x 128 x 128 GEMM on v_wmma_f32_16x16x32_bf16 with a fused
//            mobius/logmap epilogue whose scalar chain runs once per wave
//            (lane-parallel over the 16 tile rows) instead of 8x redundantly.
//  kernel 2: attention softmax computed lane-parallel (1 sigmoid per lane),
//            weights broadcast by ds_bpermute during aggregation.
// ---------------------------------------------------------------------------

typedef __attribute__((ext_vector_type(16))) __bf16 v16bf;
typedef __attribute__((ext_vector_type(2)))  __bf16 v2bf;
typedef __attribute__((ext_vector_type(16))) float  v16f;
typedef __attribute__((ext_vector_type(2)))  float  v2f;
typedef __attribute__((ext_vector_type(8)))  float  v8f;

union Frag {
    unsigned int u[8];
    v16bf        v;
};

union Cvt2 {
    v2bf         b;
    unsigned int u;
};

#define MIN_NORM_F 1e-15f
#define PROJ_EPS_F 4e-3f

// ---- helpers --------------------------------------------------------------

__device__ __forceinline__ unsigned int cvt_pk_bf16(float a, float b) {
    v2f f;
    f[0] = a; f[1] = b;
    Cvt2 c;
    c.b = __builtin_convertvector(f, v2bf);   // RNE fptrunc -> v_cvt_pk_bf16_f32
    return c.u;
}

__device__ __forceinline__ float shfl_xor_f(float v, int mask) {
    int lane = (int)(threadIdx.x & 31u);
    int src  = (lane ^ mask) << 2;
    return __int_as_float(__builtin_amdgcn_ds_bpermute(src, __float_as_int(v)));
}

__device__ __forceinline__ float shfl_idx_f(float v, int srcLane) {
    return __int_as_float(__builtin_amdgcn_ds_bpermute(srcLane << 2, __float_as_int(v)));
}

__device__ __forceinline__ float artanh_c(float x) {
    x = fminf(fmaxf(x, -1.0f + 1e-7f), 1.0f - 1e-7f);
    return 0.5f * __logf((1.0f + x) / (1.0f - x));
}

__device__ __forceinline__ float tanh_pos(float x) {
    // valid/stable for x >= 0 (e <= 1, no overflow)
    float e = __expf(-2.0f * x);
    return (1.0f - e) / (1.0f + e);
}

__device__ __forceinline__ float sigmoid_f(float x) {
    return 1.0f / (1.0f + __expf(-x));
}

// ---- kernel 0: hyp_bias = proj(expmap0(bias,c), c); also store |y|^2 ------

__global__ void __launch_bounds__(128)
prep_bias_kernel(const float* __restrict__ bias,
                 const float* __restrict__ c_in_p,
                 float* __restrict__ ws_bias /* [0..127]=y, [128]=|y|^2 */) {
    __shared__ float red[128];
    int tid = threadIdx.x;
    float b = bias[tid];
    red[tid] = b * b;
    __syncthreads();
    #pragma unroll
    for (int s = 64; s > 0; s >>= 1) {
        if (tid < s) red[tid] += red[tid + s];
        __syncthreads();
    }
    float n2 = red[0];
    float c  = c_in_p[0];
    float sc = sqrtf(c);
    float mx = (1.0f - PROJ_EPS_F) / sc;
    float un = fmaxf(sqrtf(n2), MIN_NORM_F);
    float scale = tanh_pos(sc * un) / (sc * un);   // expmap0 scale
    float yn = fabsf(scale) * un;
    if (yn > mx) { scale *= mx / yn; yn = mx; }    // proj
    ws_bias[tid] = scale * b;
    if (tid == 0) ws_bias[128] = yn * yn;
}

// ---- kernel 1: fused hyp_linear + logmap0 + attention-logit ----------------
//
// Rows 0..N-1 are self_emb, rows N..total-1 are neigh_emb.
// One 256-thread block = 8 waves; one wave owns a 16-row tile and produces
// a 16x128 tangent tile + 16 logit scalars.

__global__ void __launch_bounds__(256)
hyp_linear_kernel(const float* __restrict__ self_emb,
                  const float* __restrict__ neigh_emb,
                  const float* __restrict__ weight,   // [128 x 128] row-major (dout,din)
                  const float* __restrict__ att_w,    // [256]
                  const float* __restrict__ c_in_p,
                  const float* __restrict__ ws_bias,  // [0..127]=y, [128]=y2
                  float* __restrict__ tanbuf,         // [total_rows x 128]
                  float* __restrict__ logitbuf,       // [total_rows]
                  int N, int total_rows) {
    __shared__ __align__(16) unsigned int wt[128 * 64]; // 128x128 bf16, k-major per row
    __shared__ float ylds[128];
    __shared__ float awlds[256];
    __shared__ float y2lds;

    const int tid = threadIdx.x;

    // ---- stage weight (fp32 -> bf16) into LDS: element nn*128+k ----
    {
        const float2* w2 = reinterpret_cast<const float2*>(weight);
        #pragma unroll
        for (int i = 0; i < 32; ++i) {
            int p = tid * 32 + i;            // pair index, flat order matches [nn][k]
            float2 f = w2[p];
            wt[p] = cvt_pk_bf16(f.x, f.y);
        }
        if (tid < 128) ylds[tid] = ws_bias[tid];
        awlds[tid] = att_w[tid];
        if (tid == 0) y2lds = ws_bias[128];
    }
    __syncthreads();

    const int wave = tid >> 5;
    const int lane = tid & 31;
    const int tile = blockIdx.x * 8 + wave;
    if (tile * 16 >= total_rows) return;

    const int col_l = lane & 15;   // column within a 16-wide N group
    const int half  = lane >> 4;   // lane half selects K sub-runs / row half

    // Row this lane loads A-fragments for:
    int g_row = tile * 16 + col_l;
    if (g_row >= total_rows) g_row = total_rows - 1;
    const float* xrow = (g_row < N) ? (self_emb + (size_t)g_row * 128)
                                    : (neigh_emb + (size_t)(g_row - N) * 128);

    const float c   = c_in_p[0];
    const float sc  = sqrtf(c);
    const float mxn_cap = (1.0f - PROJ_EPS_F) / sc;
    const float y2  = y2lds;

    // per-lane column-constants
    float yreg[8], wreg_s[8], wreg_n[8];
    #pragma unroll
    for (int jj = 0; jj < 8; ++jj) {
        yreg[jj]   = ylds[jj * 16 + col_l];
        wreg_s[jj] = awlds[jj * 16 + col_l];          // att_w[0:128]
        wreg_n[jj] = awlds[128 + jj * 16 + col_l];    // att_w[128:256]
    }

    // ---- GEMM: C[16x128] = X[16x128] * W^T, bf16 WMMA, f32 accumulate ----
    v8f acc[8];
    #pragma unroll
    for (int jj = 0; jj < 8; ++jj) acc[jj] = (v8f){};

    float x2v = 0.0f; // per-lane partial |x_row|^2 (halves combined below)

    #pragma unroll
    for (int kk = 0; kk < 4; ++kk) {
        const int kb = kk * 32 + half * 8;
        const float4 f0 = *reinterpret_cast<const float4*>(xrow + kb);
        const float4 f1 = *reinterpret_cast<const float4*>(xrow + kb + 4);
        const float4 f2 = *reinterpret_cast<const float4*>(xrow + kb + 16);
        const float4 f3 = *reinterpret_cast<const float4*>(xrow + kb + 20);

        x2v += f0.x*f0.x + f0.y*f0.y + f0.z*f0.z + f0.w*f0.w
             + f1.x*f1.x + f1.y*f1.y + f1.z*f1.z + f1.w*f1.w
             + f2.x*f2.x + f2.y*f2.y + f2.z*f2.z + f2.w*f2.w
             + f3.x*f3.x + f3.y*f3.y + f3.z*f3.z + f3.w*f3.w;

        // whole 16-element fragment converted at once -> packed cvt ops
        v16f af;
        af[0]  = f0.x; af[1]  = f0.y; af[2]  = f0.z; af[3]  = f0.w;
        af[4]  = f1.x; af[5]  = f1.y; af[6]  = f1.z; af[7]  = f1.w;
        af[8]  = f2.x; af[9]  = f2.y; af[10] = f2.z; af[11] = f2.w;
        af[12] = f3.x; af[13] = f3.y; af[14] = f3.z; af[15] = f3.w;
        Frag a;
        a.v = __builtin_convertvector(af, v16bf);

        #pragma unroll
        for (int jj = 0; jj < 8; ++jj) {
            const unsigned eoff = (unsigned)(jj * 16 + col_l) * 64u
                                + (unsigned)kk * 16u + (unsigned)half * 4u;
            const uint4* bp = reinterpret_cast<const uint4*>(&wt[eoff]);
            uint4 q0 = bp[0];
            uint4 q1 = bp[2];
            Frag b;
            b.u[0] = q0.x; b.u[1] = q0.y; b.u[2] = q0.z; b.u[3] = q0.w;
            b.u[4] = q1.x; b.u[5] = q1.y; b.u[6] = q1.z; b.u[7] = q1.w;
            acc[jj] = __builtin_amdgcn_wmma_f32_16x16x32_bf16(
                          false, a.v, false, b.v, (short)0, acc[jj], false, false);
        }
    }
    // combine K halves of |x|^2 : lane l and l^16 each hold half of row (l&15)
    x2v += shfl_xor_f(x2v, 16);

    // ---- phase 1: per-row norm / dot reductions; distribute one row per lane
    // After the butterflies for row r, keep the result only in lanes with
    // (col_l & 7) == r.  Lane (16h + cl) then owns row (cl & 7) + 8h.
    float rs2v = 0.0f, dxyv = 0.0f;
    #pragma unroll
    for (int r = 0; r < 8; ++r) {
        float rs2 = 0.0f, dxy = 0.0f;
        #pragma unroll
        for (int jj = 0; jj < 8; ++jj) {
            float v = acc[jj][r];
            rs2 += v * v;                    // |mx_row|^2 partial
            dxy += v * yreg[jj];             // <mx_row, y> partial
        }
        #pragma unroll
        for (int m = 1; m < 16; m <<= 1) {
            rs2 += shfl_xor_f(rs2, m);
            dxy += shfl_xor_f(dxy, m);
        }
        if ((col_l & 7) == r) { rs2v = rs2; dxyv = dxy; }
    }
    const int Ma = (col_l & 7) + half * 8;   // row this lane owns
    const float x2m = shfl_idx_f(x2v, Ma);   // lane M (0..15) holds row M's |x|^2

    // ---- scalar chain: executed ONCE per wave, lane-parallel over 16 rows ---
    // mobius_matvec scale: res = s1 * mx
    float xn  = fmaxf(sqrtf(x2m),  MIN_NORM_F);
    float mxn = fmaxf(sqrtf(rs2v), MIN_NORM_F);
    float s1  = tanh_pos((mxn / xn) * artanh_c(sc * xn)) / (mxn * sc);
    if (rs2v == 0.0f) s1 = 0.0f;             // zero_rows
    // proj(h)
    float hn = fabsf(s1) * mxn;
    if (hn > mxn_cap) { s1 *= mxn_cap / hn; hn = mxn_cap; }
    const float h2 = hn * hn;
    const float xy = s1 * dxyv;              // <h, y>
    // mobius_add(h, y): num = alpha*h + beta*y ; den
    const float alpha = 1.0f + 2.0f * c * xy + c * y2;
    const float beta  = 1.0f - c * h2;
    const float den   = fmaxf(1.0f + 2.0f * c * xy + c * c * h2 * y2, MIN_NORM_F);
    const float num2  = alpha * alpha * h2 + 2.0f * alpha * beta * xy + beta * beta * y2;
    // proj then logmap0, folded into two scalars
    float pn = sqrtf(fmaxf(num2, 0.0f)) / den;
    float gamma = 1.0f;
    if (pn > mxn_cap) { gamma = mxn_cap / pn; pn = mxn_cap; }
    const float lam = artanh_c(sc * pn) / (sc * fmaxf(pn, MIN_NORM_F));
    const float sAv = lam * gamma * alpha * s1 / den;  // multiplies mx element
    const float sBv = lam * gamma * beta / den;        // multiplies y element

    // ---- phase 2: broadcast per-row scalars, emit tangents + logits --------
    #pragma unroll
    for (int r = 0; r < 8; ++r) {
        const int M = r + half * 8;                    // row within tile
        const float sA = shfl_idx_f(sAv, half * 16 + r);
        const float sB = shfl_idx_f(sBv, half * 16 + r);

        const int grow = tile * 16 + M;
        const bool row_self = (grow < N);

        float lpart = 0.0f;
        float tanv[8];
        #pragma unroll
        for (int jj = 0; jj < 8; ++jj) {
            tanv[jj] = sA * acc[jj][r] + sB * yreg[jj];
            lpart += tanv[jj] * (row_self ? wreg_s[jj] : wreg_n[jj]);
        }
        #pragma unroll
        for (int m = 1; m < 16; m <<= 1) lpart += shfl_xor_f(lpart, m);

        if (grow < total_rows) {
            float* trow = tanbuf + (size_t)grow * 128;
            #pragma unroll
            for (int jj = 0; jj < 8; ++jj)
                trow[jj * 16 + col_l] = tanv[jj];       // coalesced across lanes
            if (col_l == 0) logitbuf[grow] = lpart;
        }
    }
}

// ---- kernel 2: attention softmax + aggregation + relu + expmap0 + proj ----
// one wave per node; requires nneigh <= 16 (reference uses 16).

__global__ void __launch_bounds__(256)
aggregate_kernel(const float* __restrict__ tanbuf,
                 const float* __restrict__ logitbuf,
                 const float* __restrict__ att_b_p,
                 const float* __restrict__ c_out_p,
                 float* __restrict__ out,
                 int N, int nneigh) {
    const int wave = threadIdx.x >> 5;
    const int lane = threadIdx.x & 31;
    const int i = blockIdx.x * 8 + wave;
    if (i >= N) return;

    const float att_b = att_b_p[0];
    const float co = c_out_p[0];
    const float sc = sqrtf(co);
    const float mxn_cap = (1.0f - PROJ_EPS_F) / sc;

    const float sl = logitbuf[i];
    const float* nl = logitbuf + N + (size_t)i * nneigh;

    // lane-parallel softmax(sigmoid(logit)): lane j handles neighbor j
    const int jl = lane & 15;
    float a = -1e30f;
    if (jl < nneigh) a = sigmoid_f(sl + nl[jl] + att_b);
    float m = a;
    #pragma unroll
    for (int mm = 1; mm < 16; mm <<= 1) m = fmaxf(m, shfl_xor_f(m, mm));
    float e = (jl < nneigh) ? __expf(a - m) : 0.0f;
    float ssum = e;
    #pragma unroll
    for (int mm = 1; mm < 16; mm <<= 1) ssum += shfl_xor_f(ssum, mm);
    const float inv = 1.0f / ssum;

    // aggregation: weights broadcast from lane j
    const int d0 = lane * 4;
    float4 agg = make_float4(0.f, 0.f, 0.f, 0.f);
    for (int j = 0; j < nneigh; ++j) {
        const float w = shfl_idx_f(e, j) * inv;
        const float4 nt = *reinterpret_cast<const float4*>(
            tanbuf + ((size_t)N + (size_t)i * nneigh + j) * 128 + d0);
        agg.x += w * nt.x; agg.y += w * nt.y; agg.z += w * nt.z; agg.w += w * nt.w;
    }

    const float4 se = *reinterpret_cast<const float4*>(tanbuf + (size_t)i * 128 + d0);
    float4 u;
    u.x = fmaxf(0.5f * (se.x + agg.x), 0.0f);
    u.y = fmaxf(0.5f * (se.y + agg.y), 0.0f);
    u.z = fmaxf(0.5f * (se.z + agg.z), 0.0f);
    u.w = fmaxf(0.5f * (se.w + agg.w), 0.0f);

    float s = u.x*u.x + u.y*u.y + u.z*u.z + u.w*u.w;
    #pragma unroll
    for (int mm = 1; mm < 32; mm <<= 1) s += shfl_xor_f(s, mm);

    float un = fmaxf(sqrtf(s), MIN_NORM_F);
    float scale = tanh_pos(sc * un) / (sc * un);   // expmap0
    float rn = fabsf(scale) * un;
    if (rn > mxn_cap) scale *= mxn_cap / rn;       // proj

    float4 res = make_float4(scale * u.x, scale * u.y, scale * u.z, scale * u.w);
    *reinterpret_cast<float4*>(out + (size_t)i * 128 + d0) = res;
}

// ---------------------------------------------------------------------------

extern "C" void kernel_launch(void* const* d_in, const int* in_sizes, int n_in,
                              void* d_out, int out_size, void* d_ws, size_t ws_size,
                              hipStream_t stream) {
    const float* self_emb  = (const float*)d_in[0];
    const float* neigh_emb = (const float*)d_in[1];
    const float* weight    = (const float*)d_in[2];
    const float* bias      = (const float*)d_in[3];
    const float* att_w     = (const float*)d_in[4];
    const float* att_b     = (const float*)d_in[5];
    const float* c_in      = (const float*)d_in[6];
    const float* c_out     = (const float*)d_in[7];
    float* out = (float*)d_out;

    const int D = 128;
    const int N = in_sizes[0] / D;                 // 50000
    const int nneigh = in_sizes[1] / in_sizes[0];  // 16
    const int total_rows = N + N * nneigh;         // 850000

    // workspace layout (floats):
    //   [0 .. 128]        : hyp_bias (128) + |y|^2 (1)   (padded to 256)
    //   [256 ..)          : tangent buffer [total_rows x 128]
    //   then              : logits [total_rows]
    float* wsf      = (float*)d_ws;
    float* ws_bias  = wsf;
    float* tanbuf   = wsf + 256;
    float* logitbuf = tanbuf + (size_t)total_rows * D;

    prep_bias_kernel<<<1, 128, 0, stream>>>(bias, c_in, ws_bias);

    const int total_tiles = (total_rows + 15) / 16;
    const int blocks1 = (total_tiles + 7) / 8;
    hyp_linear_kernel<<<blocks1, 256, 0, stream>>>(
        self_emb, neigh_emb, weight, att_w, c_in, ws_bias,
        tanbuf, logitbuf, N, total_rows);

    const int blocks2 = (N + 7) / 8;
    aggregate_kernel<<<blocks2, 256, 0, stream>>>(
        tanbuf, logitbuf, att_b, c_out, out, N, nneigh);
}
